// PANet_21199958573586
// MI455X (gfx1250) — matrix operand order repaired and checked by reading, not compiled, not run on hardware
//
#include <hip/hip_runtime.h>

typedef __attribute__((ext_vector_type(16))) _Float16 v16h;
typedef __attribute__((ext_vector_type(8)))  _Float16 v8h;
typedef __attribute__((ext_vector_type(8)))  float    v8f;
typedef __attribute__((ext_vector_type(4)))  unsigned u32x4;
typedef __attribute__((ext_vector_type(8)))  int      i32x8;
typedef __attribute__((ext_vector_type(4)))  int      i32x4;

#define BB   2
#define CHN  32
#define HH   256
#define WW   512
#define DD   192
#define PADL 256
#define WPAD 1024
#define HW   (HH * WW)

#define FEAT_ELEMS ((size_t)BB * HH * WPAD)   // elems per feature buffer
// f16 weight region (element offsets)
#define WDH_OFF 0
#define WDF_OFF 36864
#define W1H_OFF (WDF_OFF + 36864)
#define W2H_OFF (W1H_OFF + 2048)
#define W3H_OFF (W2H_OFF + 2048)

#if defined(__HIP_DEVICE_COMPILE__) && __has_builtin(__builtin_amdgcn_tensor_load_to_lds)
#define HAVE_TDM 1
#else
#define HAVE_TDM 0
#endif

__device__ __forceinline__ v8f vzero8() {
    v8f z;
#pragma unroll
    for (int r = 0; r < 8; r++) z[r] = 0.0f;
    return z;
}

__device__ __forceinline__ v8f wmma16(v16h a, v16h b, v8f c) {
    return __builtin_amdgcn_wmma_f32_16x16x32_f16(false, a, false, b, (short)0, c,
                                                  false, false);
}

// A fragment (16x32 f16): k=16*hf+e (p[0..15]).
// Two aligned 16B loads (LDS or global).
__device__ __forceinline__ v16h load_afrag(const _Float16* p) {
    const v8h* q = (const v8h*)p;
    v8h lo = q[0], hi = q[1];
    v16h a;
#pragma unroll
    for (int i = 0; i < 8; i++) { a[i] = lo[i]; a[i + 8] = hi[i]; }
    return a;
}

// B fragment from 16 contiguous halfs, 16B-aligned source (LDS scratch).
__device__ __forceinline__ v16h load_bfrag(const _Float16* p) {
    const v8h* q = (const v8h*)p;
    v8h lo = q[0], hi = q[1];
    v16h b;
#pragma unroll
    for (int i = 0; i < 8; i++) { b[i] = lo[i]; b[i + 8] = hi[i]; }
    return b;
}

// B fragment from 16 contiguous halfs, arbitrary (2B) alignment (global).
__device__ __forceinline__ v16h load_bfrag_u(const _Float16* p) {
    v8h lo, hi;
    __builtin_memcpy(&lo, p, sizeof(v8h));
    __builtin_memcpy(&hi, p + 8, sizeof(v8h));
    v16h b;
#pragma unroll
    for (int i = 0; i < 8; i++) { b[i] = lo[i]; b[i + 8] = hi[i]; }
    return b;
}

// ---------------------------------------------------------------------------
// Kernel 0: one-shot f32->f16 weight conversion (wd normal + d-reversed,
// w1, w2, w3 zero-padded to 16x32).  41472 threads.
// ---------------------------------------------------------------------------
__global__ __launch_bounds__(256) void prep_kernel(
    const float* __restrict__ wd, const float* __restrict__ w1,
    const float* __restrict__ w2, const float* __restrict__ w3,
    _Float16* __restrict__ wdH, _Float16* __restrict__ wdF,
    _Float16* __restrict__ w1H, _Float16* __restrict__ w2H,
    _Float16* __restrict__ w3H)
{
    int i = blockIdx.x * 256 + threadIdx.x;
    if (i < 36864) {
        int o = i / DD, j = i - o * DD;
        wdH[i] = (_Float16)wd[i];
        wdF[i] = (_Float16)wd[o * DD + (DD - 1 - j)];
        return;
    }
    i -= 36864;
    if (i < 2048) { w1H[i] = (_Float16)w1[i]; return; }
    i -= 2048;
    if (i < 2048) { w2H[i] = (_Float16)w2[i]; return; }
    i -= 2048;
    if (i < 512)  { w3H[i] = (i < 32) ? (_Float16)w3[i] : (_Float16)0.0f; }
}

// ---------------------------------------------------------------------------
// Kernel 1: zero the pads of the f32 and f16 feature rows.
// 2 feats * 512 rows * 512 pad cols = 524288 threads.
// ---------------------------------------------------------------------------
__global__ __launch_bounds__(256) void pad_kernel(
    float* __restrict__ featX,  float* __restrict__ featY,
    _Float16* __restrict__ featXh, _Float16* __restrict__ featYh)
{
    unsigned i = blockIdx.x * 256u + threadIdx.x;
    int p   = i & 511;
    int row = (i >> 9) & 511;          // BB*HH = 512 rows
    int f   = i >> 18;
    int wp  = (p < PADL) ? p : p + WW; // [0,256) U [768,1024)
    size_t idx = (size_t)row * WPAD + wp;
    (f ? featY  : featX )[idx] = 0.0f;
    (f ? featYh : featXh)[idx] = (_Float16)0.0f;
}

// ---------------------------------------------------------------------------
// Kernel 2: WMMA encoder.  One wave == 16 pixels of one row of one image.
// 9 WMMAs; activations bounce through a per-wave LDS scratch [n][72].
// Writes features as f32 (softmax scale) and f16 (B fragments).
// ---------------------------------------------------------------------------
__global__ __launch_bounds__(128) void encoder_wmma(
    const float* __restrict__ x,  const float* __restrict__ y,
    const float* __restrict__ b1, const float* __restrict__ b2,
    const float* __restrict__ b3,
    const _Float16* __restrict__ w1H, const _Float16* __restrict__ w2H,
    const _Float16* __restrict__ w3H,
    float* __restrict__ featX, float* __restrict__ featY,
    _Float16* __restrict__ featXh, _Float16* __restrict__ featYh)
{
    __shared__ __align__(16) _Float16 scr[4][16 * 72];

    int wave = threadIdx.x >> 5, lane = threadIdx.x & 31;
    int n = lane & 15, hf = lane >> 4;
    _Float16* S = scr[wave];

    int bid = blockIdx.x;              // 2 imgs * 2 b * 256 h * 8 groups
    int tg  = bid & 7;
    int h   = (bid >> 3) & (HH - 1);
    int b   = (bid >> 11) & 1;
    int img = (bid >> 12) & 1;
    int w0  = (tg * 4 + wave) * 16;

    const float* in = (img ? y : x) + ((size_t)(b * CHN) * HH + h) * WW + w0 + n;

    // Layer-0 B fragment: k = channel = 16*hf + e, with input relu.
    v16h b0;
#pragma unroll
    for (int e = 0; e < 16; e++)
        b0[e] = (_Float16)fmaxf(in[(size_t)(16 * hf + e) * HW], 0.0f);

    // Layer 1: (64x32) @ (32x16) -> 4 o-tiles.
    v8f a1[4];
#pragma unroll
    for (int t = 0; t < 4; t++)
        a1[t] = wmma16(load_afrag(w1H + (t * 16 + n) * 32 + hf * 16), b0, vzero8());
#pragma unroll
    for (int t = 0; t < 4; t++) {
        const float* pb = b1 + t * 16 + hf * 8;
        v8h pk;
#pragma unroll
        for (int r = 0; r < 8; r++) pk[r] = (_Float16)fmaxf(a1[t][r] + pb[r], 0.0f);
        *(v8h*)(S + n * 72 + t * 16 + hf * 8) = pk;   // h1[m=16t+8hf+r][n]
    }

    // Layer 2: (32x64) @ (64x16) -> 2 o-tiles x 2 k-chunks.
    v8f a2[2] = { vzero8(), vzero8() };
#pragma unroll
    for (int kc = 0; kc < 2; kc++) {
        v16h bf = load_bfrag(S + n * 72 + kc * 32 + hf * 16);
#pragma unroll
        for (int t = 0; t < 2; t++)
            a2[t] = wmma16(load_afrag(w2H + (t * 16 + n) * 64 + kc * 32 + hf * 16),
                           bf, a2[t]);
    }
#pragma unroll
    for (int t = 0; t < 2; t++) {
        const float* pb = b2 + t * 16 + hf * 8;
        v8h pk;
#pragma unroll
        for (int r = 0; r < 8; r++) pk[r] = (_Float16)fmaxf(a2[t][r] + pb[r], 0.0f);
        *(v8h*)(S + n * 72 + t * 16 + hf * 8) = pk;   // h2
    }

    // Layer 3: (16x32 zero-padded) @ (32x16); only row m=0 is real.
    v16h bf3 = load_bfrag(S + n * 72 + hf * 16);
    v8f a3 = wmma16(load_afrag(w3H + n * 32 + hf * 16), bf3, vzero8());

    if (hf == 0) {  // m=0 lives in VGPR r=0 of lanes 0..15
        float v = fmaxf(a3[0] + b3[0], 0.0f);
        size_t idx = (size_t)(b * HH + h) * WPAD + PADL + w0 + n;
        (img ? featY  : featX )[idx] = v;
        (img ? featYh : featXh)[idx] = (_Float16)v;
    }
}

// ---------------------------------------------------------------------------
// Kernel 3: fused cost-volume + 192x192 projection (WMMA) + softmax +
// expectation.  One block == half a row (16 tiles, 4 waves x 4 tiles); LDS
// staging of wd (73.7 KB, TDM when available) amortizes over 1152 WMMAs.
// ---------------------------------------------------------------------------
__global__ __launch_bounds__(128) void disp_kernel(
    const _Float16* __restrict__ wdH, const _Float16* __restrict__ wdF,
    const float* __restrict__ bd,
    const float* __restrict__ featX, const float* __restrict__ featY,
    const _Float16* __restrict__ featXh, const _Float16* __restrict__ featYh,
    float* __restrict__ out)
{
    extern __shared__ char smem[];
    _Float16* sWd = (_Float16*)smem;                       // 192*192 halfs
    float*    sBd = (float*)(smem + DD * DD * sizeof(_Float16));

    int bid  = blockIdx.x;          // 2 sides * 2 b * 256 h * 2 halves
    int half = bid & 1;
    int h    = (bid >> 1) & (HH - 1);
    int b    = (bid >> 9) & 1;
    int side = (bid >> 10) & 1;

    int wave = threadIdx.x >> 5, lane = threadIdx.x & 31;
    int n = lane & 15, hf = lane >> 4;

    const _Float16* wsrc = side ? wdF : wdH;

#if HAVE_TDM
    if (wave == 0) {
        // D#: 2D dword tensor 1024 x 18 == 73728 B, contiguous, global -> LDS.
        unsigned lds_lo = (unsigned)(unsigned long long)(const void*)sWd;
        unsigned long long ga = (unsigned long long)wsrc;
        u32x4 g0 = { 1u,                       // count=1, user descriptor
                     lds_lo,                   // lds_addr
                     (unsigned)ga,             // global_addr[31:0]
                     (unsigned)((ga >> 32) & 0x01FFFFFFu) | (2u << 30) }; // type=2
        i32x8 g1 = { (int)(2u << 16),          // data_size = 4B
                     (int)(1024u << 16),       // tensor_dim0[15:0] = 1024
                     (int)(18u << 16),         // tensor_dim0 hi=0 | tensor_dim1=18
                     (int)(1024u << 16),       // tensor_dim1 hi=0 | tile_dim0=1024
                     18,                       // tile_dim1=18 | tile_dim2=0
                     1024,                     // tensor_dim0_stride[31:0]
                     0, 0 };
        i32x4 gz = { 0, 0, 0, 0 };
#if __has_include(<hip/amd_detail/amd_gfx1250_TDM.h>)
        i32x8 gz8 = { 0, 0, 0, 0, 0, 0, 0, 0 };
        __builtin_amdgcn_tensor_load_to_lds(g0, g1, gz, gz, gz8, 0);
#else
        __builtin_amdgcn_tensor_load_to_lds(g0, g1, gz, gz, 0);
#endif
        __builtin_amdgcn_s_wait_tensorcnt(0);
    }
#else
    for (int i = threadIdx.x; i < DD * DD / 8; i += 128)
        ((v8h*)sWd)[i] = ((const v8h*)wsrc)[i];
#endif
    for (int i = threadIdx.x; i < DD; i += 128) sBd[i] = bd[i];
    __syncthreads();

    size_t rowOff = (size_t)(b * HH + h) * WPAD + PADL;
    const float*    rowX  = featX  + rowOff;
    const float*    rowY  = featY  + rowOff;
    const _Float16* rowH  = (side ? featYh : featXh) + rowOff;

    for (int tile = half * 16 + wave; tile < half * 16 + 16; tile += 4) {
        int w0 = tile * 16;
        const _Float16* srcH = rowH + w0 - (side ? (DD - 1) : 0);
        float scale = (side ? rowX : rowY)[w0 + n];

        v8f acc[12];
#pragma unroll
        for (int t = 0; t < 12; t++) acc[t] = vzero8();

        for (int kc = 0; kc < 6; kc++) {
            v16h bf = load_bfrag_u(srcH + n + kc * 32 + hf * 16);
#pragma unroll
            for (int t = 0; t < 12; t++)
                acc[t] = wmma16(
                    load_afrag(sWd + (t * 16 + n) * DD + kc * 32 + hf * 16),
                    bf, acc[t]);
        }

        // z = scale*acc + bd; softmax over o (lane half hf owns o=16t+8hf+r).
        float mx = -3.4e38f;
#pragma unroll
        for (int t = 0; t < 12; t++)
#pragma unroll
            for (int r = 0; r < 8; r++) {
                float z = fmaf(scale, acc[t][r], sBd[t * 16 + hf * 8 + r]);
                acc[t][r] = z;
                mx = fmaxf(mx, z);
            }
        mx = fmaxf(mx, __shfl_xor(mx, 16, 32));

        float s = 0.0f, sw = 0.0f;
#pragma unroll
        for (int t = 0; t < 12; t++)
#pragma unroll
            for (int r = 0; r < 8; r++) {
                float e = __expf(acc[t][r] - mx);
                s += e;
                sw = fmaf(e, (float)(t * 16 + hf * 8 + r), sw);
            }
        s  += __shfl_xor(s, 16, 32);
        sw += __shfl_xor(sw, 16, 32);

        if (hf == 0) {
            float* op = out + (size_t)side * (BB * HH * WW)
                            + (size_t)(b * HH + h) * WW + w0 + n;
            *op = sw / s;
        }
    }
}

// ---------------------------------------------------------------------------
extern "C" void kernel_launch(void* const* d_in, const int* in_sizes, int n_in,
                              void* d_out, int out_size, void* d_ws, size_t ws_size,
                              hipStream_t stream) {
    (void)in_sizes; (void)n_in; (void)out_size; (void)ws_size;
    const float* x  = (const float*)d_in[0];
    const float* y  = (const float*)d_in[1];
    const float* w1 = (const float*)d_in[2];
    const float* b1 = (const float*)d_in[3];
    const float* w2 = (const float*)d_in[4];
    const float* b2 = (const float*)d_in[5];
    const float* w3 = (const float*)d_in[6];
    const float* b3 = (const float*)d_in[7];
    const float* wd = (const float*)d_in[8];
    const float* bd = (const float*)d_in[9];

    float* featX = (float*)d_ws;
    float* featY = featX + FEAT_ELEMS;
    _Float16* featXh = (_Float16*)(featY + FEAT_ELEMS);
    _Float16* featYh = featXh + FEAT_ELEMS;
    _Float16* hreg   = featYh + FEAT_ELEMS;
    _Float16* wdH = hreg + WDH_OFF;
    _Float16* wdF = hreg + WDF_OFF;
    _Float16* w1H = hreg + W1H_OFF;
    _Float16* w2H = hreg + W2H_OFF;
    _Float16* w3H = hreg + W3H_OFF;

    prep_kernel<<<162, 256, 0, stream>>>(wd, w1, w2, w3, wdH, wdF, w1H, w2H, w3H);
    pad_kernel<<<2048, 256, 0, stream>>>(featX, featY, featXh, featYh);
    encoder_wmma<<<2 * BB * HH * 8, 128, 0, stream>>>(x, y, b1, b2, b3,
                                                      w1H, w2H, w3H,
                                                      featX, featY, featXh, featYh);

    size_t shmem = (size_t)DD * DD * sizeof(_Float16) + DD * sizeof(float);
    disp_kernel<<<2 * BB * HH * 2, 128, shmem, stream>>>(wdH, wdF, bd,
                                                         featX, featY,
                                                         featXh, featYh,
                                                         (float*)d_out);
}
